// CustomCLIP_18038862643653
// MI455X (gfx1250) — compile-verified, matrix operand order checked by reading, not run on hardware
//
#include <hip/hip_runtime.h>

#define DEV __device__ __forceinline__

typedef __bf16 bf16;
typedef __attribute__((ext_vector_type(16))) __bf16 v16bf;
typedef __attribute__((ext_vector_type(8)))  __bf16 v8bf;
typedef __attribute__((ext_vector_type(8)))  float  v8f;

#if __has_builtin(__builtin_amdgcn_global_load_async_to_lds_b128) && \
    __has_builtin(__builtin_amdgcn_s_wait_asynccnt)
#define USE_ASYNC_LDS 1
#else
#define USE_ASYNC_LDS 0
#endif

// async-to-LDS builtin takes (global v4i*, lds v4i*, imm offset, imm cpol)
typedef int v4i_vs __attribute__((vector_size(16)));
#define GPTR_V4I(p) ((__attribute__((address_space(1))) v4i_vs*)(p))
#define LPTR_V4I(p) ((__attribute__((address_space(3))) v4i_vs*)(p))

// ---------------- problem constants ----------------
constexpr int   M      = 196;          // image patches
constexpr int   BI     = 32;           // images
constexpr int   D      = 512;          // feature dim
constexpr int   NC     = 1000;         // classes
constexpr int   NP     = 4;            // prompt ensembles
constexpr int   COLS   = NP * NC;      // 4000 GEMM columns
constexpr int   BTOT   = BI * NC;      // 32000 sinkhorn problems
constexpr float U_VAL  = 1.0f / 196.0f;
constexpr float V_VAL  = 0.25f;
constexpr float THRESH = 0.01f;
constexpr int   MAXIT  = 100;
constexpr int   KSTEPS = D / 32;       // 16 k-steps of 32
constexpr int   UPD_BLOCKS = BTOT / 256;   // 125

// ---------------- helpers ----------------
DEV bf16 f2bf(float f) {                       // RNE f32 -> bf16
  unsigned u = __builtin_bit_cast(unsigned, f);
  u = (u + 0x7FFFu + ((u >> 16) & 1u)) >> 16;
  unsigned short s = (unsigned short)u;
  return __builtin_bit_cast(bf16, s);
}

DEV float waveSum(float v) {
  #pragma unroll
  for (int off = 16; off > 0; off >>= 1) v += __shfl_xor(v, off, 32);
  return v;
}

DEV v16bf loadFragA(const bf16* __restrict__ Ab, int k0, int half) {
  int koff = k0 + half * 8;
  v8bf lo = *(const v8bf*)(Ab + koff);
  v8bf hi = *(const v8bf*)(Ab + koff + 16);
  v16bf a;
  #pragma unroll
  for (int i = 0; i < 8; i++) { a[i] = lo[i]; a[i + 8] = hi[i]; }
  return a;
}

// ---------------- normalization kernels (1 wave / row, 8 waves / block) ----------------
// image_features [M][BI][D] -> l2norm rows, permuted to imf_bf16 [BI][M][D]
__global__ __launch_bounds__(256) void k_norm_img(const float* __restrict__ in,
                                                  bf16* __restrict__ outp) {
  int w    = blockIdx.x * 8 + (threadIdx.x >> 5);   // row index m*32+b, 6272 rows
  int lane = threadIdx.x & 31;
  int m = w >> 5, b = w & 31;
  const float4* src = (const float4*)(in + (size_t)w * D);
  float4 v[4]; float ss = 0.f;
  #pragma unroll
  for (int i = 0; i < 4; i++) {
    v[i] = src[lane + 32 * i];
    ss += v[i].x*v[i].x + v[i].y*v[i].y + v[i].z*v[i].z + v[i].w*v[i].w;
  }
  ss = waveSum(ss);
  float rn = 1.0f / sqrtf(ss);
  bf16* dst = outp + ((size_t)b * M + m) * D;
  #pragma unroll
  for (int i = 0; i < 4; i++) {
    int base = (lane + 32 * i) * 4;
    dst[base+0] = f2bf(v[i].x * rn); dst[base+1] = f2bf(v[i].y * rn);
    dst[base+2] = f2bf(v[i].z * rn); dst[base+3] = f2bf(v[i].w * rn);
  }
}

// text_features [COLS][D] -> l2norm rows -> tfn_bf16 [COLS][D]
__global__ __launch_bounds__(256) void k_norm_text(const float* __restrict__ in,
                                                   bf16* __restrict__ outp) {
  int w    = blockIdx.x * 8 + (threadIdx.x >> 5);   // 4000 rows
  int lane = threadIdx.x & 31;
  const float4* src = (const float4*)(in + (size_t)w * D);
  float4 v[4]; float ss = 0.f;
  #pragma unroll
  for (int i = 0; i < 4; i++) {
    v[i] = src[lane + 32 * i];
    ss += v[i].x*v[i].x + v[i].y*v[i].y + v[i].z*v[i].z + v[i].w*v[i].w;
  }
  ss = waveSum(ss);
  float rn = 1.0f / sqrtf(ss);
  bf16* dst = outp + (size_t)w * D;
  #pragma unroll
  for (int i = 0; i < 4; i++) {
    int base = (lane + 32 * i) * 4;
    dst[base+0] = f2bf(v[i].x * rn); dst[base+1] = f2bf(v[i].y * rn);
    dst[base+2] = f2bf(v[i].z * rn); dst[base+3] = f2bf(v[i].w * rn);
  }
}

// generic f32 row l2norm (used for image_feature_pool, 32 rows)
__global__ __launch_bounds__(256) void k_norm_f32(const float* __restrict__ in,
                                                  float* __restrict__ outp) {
  int w    = blockIdx.x * 8 + (threadIdx.x >> 5);
  int lane = threadIdx.x & 31;
  const float4* src = (const float4*)(in + (size_t)w * D);
  float4 v[4]; float ss = 0.f;
  #pragma unroll
  for (int i = 0; i < 4; i++) {
    v[i] = src[lane + 32 * i];
    ss += v[i].x*v[i].x + v[i].y*v[i].y + v[i].z*v[i].z + v[i].w*v[i].w;
  }
  ss = waveSum(ss);
  float rn = 1.0f / sqrtf(ss);
  float4* dst = (float4*)(outp + (size_t)w * D);
  #pragma unroll
  for (int i = 0; i < 4; i++) {
    float4 o; o.x=v[i].x*rn; o.y=v[i].y*rn; o.z=v[i].z*rn; o.w=v[i].w*rn;
    dst[lane + 32 * i] = o;
  }
}

// text pool: mean over N of raw text features, then l2norm -> txt_pool [NC][D] f32
__global__ __launch_bounds__(256) void k_text_pool(const float* __restrict__ txt,
                                                   float* __restrict__ outp) {
  int c    = blockIdx.x * 8 + (threadIdx.x >> 5);   // 1000 rows
  int lane = threadIdx.x & 31;
  const float4* s0 = (const float4*)(txt + (size_t)(0*NC + c) * D);
  const float4* s1 = (const float4*)(txt + (size_t)(1*NC + c) * D);
  const float4* s2 = (const float4*)(txt + (size_t)(2*NC + c) * D);
  const float4* s3 = (const float4*)(txt + (size_t)(3*NC + c) * D);
  float4 v[4]; float ss = 0.f;
  #pragma unroll
  for (int i = 0; i < 4; i++) {
    int idx = lane + 32 * i;
    float4 a = s0[idx], b = s1[idx], cc = s2[idx], d = s3[idx];
    float4 o;
    o.x = (a.x + b.x + cc.x + d.x) * 0.25f;
    o.y = (a.y + b.y + cc.y + d.y) * 0.25f;
    o.z = (a.z + b.z + cc.z + d.z) * 0.25f;
    o.w = (a.w + b.w + cc.w + d.w) * 0.25f;
    v[i] = o;
    ss += o.x*o.x + o.y*o.y + o.z*o.z + o.w*o.w;
  }
  ss = waveSum(ss);
  float rn = 1.0f / sqrtf(ss);
  float4* dst = (float4*)(outp + (size_t)c * D);
  #pragma unroll
  for (int i = 0; i < 4; i++) {
    float4 o; o.x=v[i].x*rn; o.y=v[i].y*rn; o.z=v[i].z*rn; o.w=v[i].w*rn;
    dst[lane + 32 * i] = o;
  }
}

// ---------------- WMMA GEMM: sim[b][m][col] = imf[b][m][:] . tfn[col][:] ----------------
// 4 waves / block; block computes 64 rows x 64 cols. The 64x32 B slab per k-step
// (4KB) is staged into LDS once per block (async double-buffered, prefetch
// distance 2) and shared by all 4 waves. Each wave: 1 m-tile x 4 col-tiles,
// A fragment register double-buffered, 4 WMMAs per k-step.
__global__ __launch_bounds__(128) void k_gemm(const bf16* __restrict__ A,   // [BI][M][D]
                                              const bf16* __restrict__ Bt,  // [COLS][D]
                                              float* __restrict__ sim) {    // [BI][M][COLS]
  __shared__ alignas(16) char smem[2][4096];     // 64 cols * 64 B (32 bf16 of K)
  int tid  = threadIdx.x;
  int lane = tid & 31;
  int wave = tid >> 5;
  int b    = blockIdx.z;
  int c0   = blockIdx.x * 64;
  int m0   = blockIdx.y * 64 + wave * 16;
  int half = lane >> 4;                          // K sub-half selector
  int l15  = lane & 15;

  int arow = m0 + l15; if (arow > M - 1) arow = M - 1;
  const bf16* Ab = A + ((size_t)b * M + arow) * D;

  // per-thread staging chunks: thread copies 16B at lin0 and 16B at lin0+2048
  int lin0 = tid * 16;
  int lin1 = lin0 + 2048;
  int col0 = c0 + (lin0 >> 6); if (col0 > COLS - 1) col0 = COLS - 1;
  int col1 = c0 + (lin1 >> 6); if (col1 > COLS - 1) col1 = COLS - 1;
  const bf16* g0 = Bt + (size_t)col0 * D + ((lin0 & 63) >> 1);
  const bf16* g1 = Bt + (size_t)col1 * D + ((lin1 & 63) >> 1);

  auto stage = [&](int buf, int k0) {
#if USE_ASYNC_LDS
    __builtin_amdgcn_global_load_async_to_lds_b128(GPTR_V4I(g0 + k0),
                                                   LPTR_V4I(&smem[buf][lin0]), 0, 0);
    __builtin_amdgcn_global_load_async_to_lds_b128(GPTR_V4I(g1 + k0),
                                                   LPTR_V4I(&smem[buf][lin1]), 0, 0);
#else
    *(uint4*)(&smem[buf][lin0]) = *(const uint4*)(g0 + k0);
    *(uint4*)(&smem[buf][lin1]) = *(const uint4*)(g1 + k0);
#endif
  };

  v8f acc[4] = {};
  // prologue: prime both buffers
  stage(0, 0);
  stage(1, 32);
  v16bf a_cur = loadFragA(Ab, 0, half);

  for (int i = 0; i < KSTEPS; i++) {
    int buf = i & 1;
#if USE_ASYNC_LDS
    if (i == KSTEPS - 1) __builtin_amdgcn_s_wait_asynccnt(0);
    else                 __builtin_amdgcn_s_wait_asynccnt(2);   // 2 copies of step i+1 in flight
#endif
    __syncthreads();                              // buf data visible to all waves
    v16bf bb[4];
    #pragma unroll
    for (int q = 0; q < 4; q++) {
      const char* sb = &smem[buf][(q * 16 + l15) * 64 + half * 16];
      v8bf lo = *(const v8bf*)(sb);
      v8bf hi = *(const v8bf*)(sb + 32);
      #pragma unroll
      for (int e = 0; e < 8; e++) { bb[q][e] = lo[e]; bb[q][e + 8] = hi[e]; }
    }
    __syncthreads();                              // all waves done reading buf
    if (i + 2 < KSTEPS) stage(buf, (i + 2) * 32); // overwrite buf for step i+2
    v16bf a_use = a_cur;
    if (i + 1 < KSTEPS) a_cur = loadFragA(Ab, (i + 1) * 32, half);
    #pragma unroll
    for (int q = 0; q < 4; q++)
      acc[q] = __builtin_amdgcn_wmma_f32_16x16x32_bf16(
          false, a_use, false, bb[q], (short)0, acc[q], false, false);
  }

  #pragma unroll
  for (int q = 0; q < 4; q++) {
    int col = c0 + q * 16 + l15;
    if (col >= COLS) continue;
    #pragma unroll
    for (int j = 0; j < 8; j++) {
      int m = m0 + j + half * 8;
      if (m < M) sim[((size_t)b * M + m) * COLS + col] = acc[q][j];
    }
  }
}

// ---------------- Sinkhorn ----------------
__global__ __launch_bounds__(256) void k_sink_init(float* __restrict__ rbuf,
                                                   float* __restrict__ cbuf,
                                                   float* __restrict__ state) {
  int i = blockIdx.x * 256 + threadIdx.x;           // 6,400,000 total
  if (i < M * BTOT) rbuf[i] = 1.0f;
  else              cbuf[i - M * BTOT] = 1.0f;
  if (i == 0) state[0] = 1e30f;                     // err = +inf
}

__global__ __launch_bounds__(256) void k_sink_update(const float* __restrict__ sim,
                                                     float* __restrict__ rbuf,
                                                     float* __restrict__ cbuf,
                                                     const float* __restrict__ state,
                                                     float* __restrict__ partials) {
  if (state[0] < THRESH) return;                    // while-loop early exit
  int t = blockIdx.x * 256 + threadIdx.x;           // (b,c) pair, 32000 exact
  int b = t / NC, cix = t - b * NC;
  const float* S = sim + (size_t)b * M * COLS + cix;
  float c0 = cbuf[t], c1 = cbuf[BTOT + t], c2 = cbuf[2*BTOT + t], c3 = cbuf[3*BTOT + t];
  float t0 = 0.f, t1 = 0.f, t2 = 0.f, t3 = 0.f, err = 0.f;
  for (int m = 0; m < M; m++) {
    const float* Sm = S + (size_t)m * COLS;
    __builtin_prefetch(Sm + 4 * COLS, 0, 0);        // global_prefetch_b8, 4 rows ahead
    __builtin_prefetch(Sm + 4 * COLS + 2 * NC, 0, 0);
    float k0 = __expf((Sm[0]      - 1.0f) * 10.0f);
    float k1 = __expf((Sm[NC]     - 1.0f) * 10.0f);
    float k2 = __expf((Sm[2*NC]   - 1.0f) * 10.0f);
    float k3 = __expf((Sm[3*NC]   - 1.0f) * 10.0f);
    float denom = k0*c0 + k1*c1 + k2*c2 + k3*c3;
    float rn = U_VAL / denom;
    float* rp = rbuf + (size_t)m * BTOT + t;
    err += fabsf(rn - *rp);
    *rp = rn;
    t0 += k0 * rn; t1 += k1 * rn; t2 += k2 * rn; t3 += k3 * rn;
  }
  cbuf[t] = V_VAL / t0; cbuf[BTOT + t] = V_VAL / t1;
  cbuf[2*BTOT + t] = V_VAL / t2; cbuf[3*BTOT + t] = V_VAL / t3;

  __shared__ float red[256];
  red[threadIdx.x] = err;
  __syncthreads();
  #pragma unroll
  for (int s = 128; s > 0; s >>= 1) {
    if (threadIdx.x < s) red[threadIdx.x] += red[threadIdx.x + s];
    __syncthreads();
  }
  if (threadIdx.x == 0) partials[blockIdx.x] = red[0];
}

__global__ void k_sink_finalize(float* __restrict__ state,
                                const float* __restrict__ partials) {
  if (state[0] < THRESH) return;                    // already converged: keep err
  float s = 0.f;
  for (int i = 0; i < UPD_BLOCKS; i++) s += partials[i];   // fixed order: deterministic
  state[0] = s / ((float)BTOT * (float)M);
}

// ---------------- final: sim_op + pooled logits ----------------
__global__ __launch_bounds__(256) void k_final(const float* __restrict__ sim,
                                               const float* __restrict__ rbuf,
                                               const float* __restrict__ cbuf,
                                               const float* __restrict__ imgpool,
                                               const float* __restrict__ txtpool,
                                               const float* __restrict__ lsc,
                                               float* __restrict__ out) {
  int t = blockIdx.x * 256 + threadIdx.x;           // 32000 exact
  int b = t / NC, cix = t - b * NC;
  const float* S = sim + (size_t)b * M * COLS + cix;
  float c0 = cbuf[t], c1 = cbuf[BTOT + t], c2 = cbuf[2*BTOT + t], c3 = cbuf[3*BTOT + t];
  float sop = 0.f;
  for (int m = 0; m < M; m++) {
    const float* Sm = S + (size_t)m * COLS;
    __builtin_prefetch(Sm + 4 * COLS, 0, 0);
    float r = rbuf[(size_t)m * BTOT + t];
    float s0 = Sm[0], s1 = Sm[NC], s2 = Sm[2*NC], s3 = Sm[3*NC];
    float k0 = __expf((s0 - 1.0f) * 10.0f);
    float k1 = __expf((s1 - 1.0f) * 10.0f);
    float k2 = __expf((s2 - 1.0f) * 10.0f);
    float k3 = __expf((s3 - 1.0f) * 10.0f);
    sop += r * (c0*k0*s0 + c1*k1*s1 + c2*k2*s2 + c3*k3*s3);
  }
  const float4* ip = (const float4*)(imgpool + (size_t)b * D);
  const float4* tp = (const float4*)(txtpool + (size_t)cix * D);
  float dot = 0.f;
  #pragma unroll 4
  for (int i = 0; i < D / 4; i++) {
    float4 a = ip[i], q = tp[i];
    dot += a.x*q.x + a.y*q.y + a.z*q.z + a.w*q.w;
  }
  float ls = __expf(lsc[0]);
  out[t] = 0.5f * ls * sop + 0.5f * ls * dot;
}

// ---------------- launcher ----------------
extern "C" void kernel_launch(void* const* d_in, const int* in_sizes, int n_in,
                              void* d_out, int out_size, void* d_ws, size_t ws_size,
                              hipStream_t stream) {
  (void)in_sizes; (void)n_in; (void)out_size; (void)ws_size;
  const float* img      = (const float*)d_in[0];   // [196,32,512]
  const float* img_pool = (const float*)d_in[1];   // [32,512]
  const float* txt      = (const float*)d_in[2];   // [4000,512]
  const float* lsc      = (const float*)d_in[3];   // scalar
  float* out = (float*)d_out;

  char* ws = (char*)d_ws;
  constexpr size_t OFF_SIM  = 0;                                    // 100,352,000 B
  constexpr size_t OFF_R    = OFF_SIM + (size_t)BI*M*COLS*4;        // 25,088,000 B
  constexpr size_t OFF_C    = OFF_R   + (size_t)M*BTOT*4;           //    512,000 B
  constexpr size_t OFF_IMF  = OFF_C   + (size_t)NP*BTOT*4;          //  6,422,528 B
  constexpr size_t OFF_TFN  = OFF_IMF + (size_t)BI*M*D*2;           //  4,096,000 B
  constexpr size_t OFF_IPN  = OFF_TFN + (size_t)COLS*D*2;           //     65,536 B
  constexpr size_t OFF_TPN  = OFF_IPN + (size_t)BI*D*4;             //  2,048,000 B
  constexpr size_t OFF_ST   = OFF_TPN + (size_t)NC*D*4;             //        256 B
  constexpr size_t OFF_PART = OFF_ST  + 256;                        //        512 B

  float* sim   = (float*)(ws + OFF_SIM);
  float* rbuf  = (float*)(ws + OFF_R);
  float* cbuf  = (float*)(ws + OFF_C);
  bf16*  imf   = (bf16*) (ws + OFF_IMF);
  bf16*  tfn   = (bf16*) (ws + OFF_TFN);
  float* ipn   = (float*)(ws + OFF_IPN);
  float* tpn   = (float*)(ws + OFF_TPN);
  float* state = (float*)(ws + OFF_ST);
  float* parts = (float*)(ws + OFF_PART);

  // 1) normalizations (1 wave per row, 8 rows per 256-thread block; all exact fits)
  k_norm_img <<<M * BI / 8, 256, 0, stream>>>(img, imf);        // 784 blocks
  k_norm_text<<<COLS   / 8, 256, 0, stream>>>(txt, tfn);        // 500 blocks
  k_norm_f32 <<<BI     / 8, 256, 0, stream>>>(img_pool, ipn);   // 4 blocks
  k_text_pool<<<NC     / 8, 256, 0, stream>>>(txt, tpn);        // 125 blocks

  // 2) bf16 WMMA GEMM (LDS-staged, async double-buffered) -> sim [32][196][4000]
  k_gemm<<<dim3((COLS + 63) / 64, (M + 63) / 64, BI), 128, 0, stream>>>(imf, tfn, sim);

  // 3) Sinkhorn: fixed launch count, device-side convergence gate
  k_sink_init<<<(M * BTOT + NP * BTOT) / 256, 256, 0, stream>>>(rbuf, cbuf, state);
  for (int it = 0; it < MAXIT; it++) {
    k_sink_update<<<UPD_BLOCKS, 256, 0, stream>>>(sim, rbuf, cbuf, state, parts);
    k_sink_finalize<<<1, 1, 0, stream>>>(state, parts);
  }

  // 4) transport-weighted similarity + pooled logits
  k_final<<<UPD_BLOCKS, 256, 0, stream>>>(sim, rbuf, cbuf, ipn, tpn, lsc, out);
}
